// HungarianMatcher_91001767068278
// MI455X (gfx1250) — compile-verified, matrix operand order checked by reading, not run on hardware
//
#include <hip/hip_runtime.h>
#include <hip/hip_bf16.h>
#include <math.h>

// ---------------- problem constants ----------------
#define BSZ    4
#define NQ_    300
#define CCLS   80
#define N_Q    1200      // BSZ*NQ_
#define T_TGT  200
#define T_PAD  208       // 13 * 16
#define Wd     160
#define HW     25600     // 160*160
#define H2     320
#define NKCH   10
#define KCHUNK 2560      // HW / NKCH
#define KS     256       // LDS staging depth (halves); KCHUNK/KS = 10 buffers
#define MT     75        // N_Q / 16
#define NT     13        // T_PAD / 16
#define MBLK   38        // ceil(MT / 2) m-tile pairs

typedef _Float16 half4  __attribute__((ext_vector_type(4)));
typedef _Float16 half8  __attribute__((ext_vector_type(8)));
typedef _Float16 half16 __attribute__((ext_vector_type(16)));
typedef float    v8f    __attribute__((ext_vector_type(8)));

// ---------------------------------------------------------------------------
// Kernel 1: bilinear (antialiased 4-tap) downsample 320x320 -> 160x160,
// emit f16 B matrix rows [T_PAD][HW] (pad rows zero) + per-target pixel sum.
// ---------------------------------------------------------------------------
__global__ void hm_tgt_prep(const float* __restrict__ tm,
                            _Float16* __restrict__ Bm,
                            float* __restrict__ tsum) {
    const int t = blockIdx.x;
    const float wt[4] = {0.125f, 0.375f, 0.375f, 0.125f};
    const float* src = tm + (size_t)t * (H2 * H2);
    float partial = 0.f;
    for (int p = threadIdx.x; p < HW; p += 256) {
        float v = 0.f;
        if (t < T_TGT) {
            int oy = p / Wd, ox = p - oy * Wd;
            float acc = 0.f, wsum = 0.f;
#pragma unroll
            for (int dy = 0; dy < 4; ++dy) {
                int iy = 2 * oy - 1 + dy;
                if (iy < 0 || iy >= H2) continue;
                float wy = wt[dy];
#pragma unroll
                for (int dx = 0; dx < 4; ++dx) {
                    int ix = 2 * ox - 1 + dx;
                    if (ix < 0 || ix >= H2) continue;
                    float w = wy * wt[dx];
                    acc += w * src[iy * H2 + ix];
                    wsum += w;
                }
            }
            v = acc / wsum;
        }
        Bm[(size_t)t * HW + p] = (_Float16)v;
        partial += v;
    }
    __shared__ float red[256];
    red[threadIdx.x] = partial;
    __syncthreads();
    for (int s = 128; s > 0; s >>= 1) {
        if (threadIdx.x < s) red[threadIdx.x] += red[threadIdx.x + s];
        __syncthreads();
    }
    if (threadIdx.x == 0) tsum[t] = red[0];
}

// ---------------------------------------------------------------------------
// Kernel 2: per-query row: A1 = sigmoid(x) f16, A2 = (log_p - log_1mp) f16
// (with torch-BCE -100 clamps), plus rowsum(sigmoid) and rowsum(log_1mp).
// ---------------------------------------------------------------------------
__global__ void hm_pred_prep(const float* __restrict__ pm,
                             _Float16* __restrict__ A1,
                             _Float16* __restrict__ A2,
                             float* __restrict__ sumS,
                             float* __restrict__ sumL) {
    const int n = blockIdx.x;
    const float4* src = (const float4*)(pm + (size_t)n * HW);
    half4* a1 = (half4*)(A1 + (size_t)n * HW);
    half4* a2 = (half4*)(A2 + (size_t)n * HW);
    float ps = 0.f, pl = 0.f;
    for (int i = threadIdx.x; i < HW / 4; i += 256) {
        float4 x = src[i];
        float xs[4] = {x.x, x.y, x.z, x.w};
        half4 h1, h2;
#pragma unroll
        for (int j = 0; j < 4; ++j) {
            float v  = xs[j];
            float s  = 1.f / (1.f + __expf(-v));
            float lp = fmaxf(-log1pf(__expf(-v)), -100.f);  // log(p)   clamped
            float lm = fmaxf(-log1pf(__expf(v)),  -100.f);  // log(1-p) clamped
            h1[j] = (_Float16)s;
            h2[j] = (_Float16)(lp - lm);
            ps += s;
            pl += lm;
        }
        a1[i] = h1;
        a2[i] = h2;
    }
    __shared__ float rs[256], rl[256];
    rs[threadIdx.x] = ps;
    rl[threadIdx.x] = pl;
    __syncthreads();
    for (int s = 128; s > 0; s >>= 1) {
        if (threadIdx.x < s) {
            rs[threadIdx.x] += rs[threadIdx.x + s];
            rl[threadIdx.x] += rl[threadIdx.x + s];
        }
        __syncthreads();
    }
    if (threadIdx.x == 0) { sumS[n] = rs[0]; sumL[n] = rl[0]; }
}

// ---------------------------------------------------------------------------
// Kernel 3: class / bbox-L1 / GIoU base cost -> d_out
// ---------------------------------------------------------------------------
__global__ void hm_base_cost(const float* __restrict__ logits,
                             const float* __restrict__ pboxes,
                             const int* __restrict__ tids,
                             const float* __restrict__ tboxes,
                             float* __restrict__ out) {
    int idx = blockIdx.x * blockDim.x + threadIdx.x;
    if (idx >= N_Q * T_TGT) return;
    int nq = idx / T_TGT, t = idx - nq * T_TGT;

    float pcx = pboxes[nq * 4 + 0], pcy = pboxes[nq * 4 + 1];
    float pw  = pboxes[nq * 4 + 2], ph  = pboxes[nq * 4 + 3];
    float tcx = tboxes[t * 4 + 0],  tcy = tboxes[t * 4 + 1];
    float tw  = tboxes[t * 4 + 2],  th  = tboxes[t * 4 + 3];

    float cbbox = fabsf(pcx - tcx) + fabsf(pcy - tcy) + fabsf(pw - tw) + fabsf(ph - th);

    float ax0 = pcx - 0.5f * pw, ay0 = pcy - 0.5f * ph;
    float ax1 = pcx + 0.5f * pw, ay1 = pcy + 0.5f * ph;
    float bx0 = tcx - 0.5f * tw, by0 = tcy - 0.5f * th;
    float bx1 = tcx + 0.5f * tw, by1 = tcy + 0.5f * th;

    float area_a = (ax1 - ax0) * (ay1 - ay0);
    float area_b = (bx1 - bx0) * (by1 - by0);
    float iw = fmaxf(0.f, fminf(ax1, bx1) - fmaxf(ax0, bx0));
    float ih = fmaxf(0.f, fminf(ay1, by1) - fmaxf(ay0, by0));
    float inter = iw * ih;
    float uni = area_a + area_b - inter;
    float iou = inter / uni;
    float cw = fmaxf(0.f, fmaxf(ax1, bx1) - fminf(ax0, bx0));
    float ch = fmaxf(0.f, fmaxf(ay1, by1) - fminf(ay0, by0));
    float areac = cw * ch;
    float giou = iou - (areac - uni) / areac;

    float lg = logits[(size_t)nq * CCLS + tids[t]];
    float cclass = -(1.f / (1.f + __expf(-lg)));

    out[idx] = 5.0f * cbbox + 2.0f * (-giou) + 2.0f * cclass;
}

// ---------------------------------------------------------------------------
// Kernel 4: WMMA GEMM. One wave computes TWO 16x16 m-tiles for a K-chunk
// (both A1/dice and A2/BCE accumulators), sharing a B tile that is staged
// into LDS with double-buffered GLOBAL_LOAD_ASYNC_TO_LDS_B128 (ASYNCcnt).
// blockDim == 32 (one wave): producer == consumer, so synchronization is
// only s_wait_asynccnt (async loads complete in order). Each async op
// copies one 512B B-row (32 lanes x 16B).
//
// NOTE: &bsh is passed into the asm as an (unreferenced) operand so the
// LDS array escapes -- otherwise LLVM proves "no stores to bsh" and folds
// the ds_load fragments to poison (observed in the previous round).
//
// Fragment addressing per CDNA5 16-bit 16x32 layout:
//   lanes 0-15: row m0+lane, K = {k..k+7} and {k+16..k+23}
//   lanes 16-31: row m0+lane-16, K = {k+8..k+15} and {k+24..k+31}
// ---------------------------------------------------------------------------
__device__ __forceinline__ half16 hm_ldfrag(const _Float16* p) {
    half8 lo = *(const half8*)(p);
    half8 hi = *(const half8*)(p + 16);
    half16 r;
#pragma unroll
    for (int i = 0; i < 8; ++i) { r[i] = lo[i]; r[i + 8] = hi[i]; }
    return r;
}

__global__ void __launch_bounds__(32)
hm_mask_gemm(const _Float16* __restrict__ A1,
             const _Float16* __restrict__ A2,
             const _Float16* __restrict__ Bm,
             float* __restrict__ accb) {
    __shared__ _Float16 bsh[2][16][KS];   // 2 x 8KB double buffer, LDS offset 0
    const void* bshp = (const void*)&bsh[0][0][0];   // escape handle for asm

    const int lane = threadIdx.x;
    const int mt0  = blockIdx.x * 2;
    const bool m1ok = (mt0 + 1) < MT;
    const int mt1  = m1ok ? (mt0 + 1) : (MT - 1);
    const int n0   = blockIdx.y * 16;
    const int kbase = blockIdx.z * KCHUNK;
    const int rsel = lane & 15;
    const int hi   = lane >> 4;

    const _Float16* a1p0 = A1 + (size_t)(mt0 * 16 + rsel) * HW + kbase + hi * 8;
    const _Float16* a2p0 = A2 + (size_t)(mt0 * 16 + rsel) * HW + kbase + hi * 8;
    const _Float16* a1p1 = A1 + (size_t)(mt1 * 16 + rsel) * HW + kbase + hi * 8;
    const _Float16* a2p1 = A2 + (size_t)(mt1 * 16 + rsel) * HW + kbase + hi * 8;

    // stage B rows [n0..n0+15], k in [kbase+kc, kbase+kc+KS) into buffer `buf`
    const char* gstage = (const char*)(Bm + (size_t)n0 * HW + kbase) + lane * 16;
    const unsigned lstage = (unsigned)lane * 16u;

#define HM_STAGE(buf, kc)                                                     \
    do {                                                                      \
        _Pragma("unroll")                                                     \
        for (int r = 0; r < 16; ++r) {                                        \
            unsigned loff = lstage + (unsigned)(buf) * (16u * KS * 2u) +      \
                            (unsigned)r * (KS * 2u);                          \
            unsigned long long ga = (unsigned long long)(gstage +             \
                            (size_t)r * HW * 2 + (size_t)(kc) * 2);          \
            asm volatile("global_load_async_to_lds_b128 %0, %1, off"          \
                         :: "v"(loff), "v"(ga), "s"(bshp) : "memory");        \
        }                                                                     \
    } while (0)

    v8f acc1a = {}, acc2a = {}, acc1b = {}, acc2b = {};

    HM_STAGE(0, 0);
    for (int c = 0; c < KCHUNK / KS; ++c) {
        if (c + 1 < KCHUNK / KS) {
            HM_STAGE((c + 1) & 1, (c + 1) * KS);
            asm volatile("s_wait_asynccnt 16" :: "s"(bshp) : "memory");
        } else {
            asm volatile("s_wait_asynccnt 0" :: "s"(bshp) : "memory");
        }
        const int buf = c & 1;
        const int kc  = c * KS;
#pragma unroll 2
        for (int kk = 0; kk < KS; kk += 32) {
            const int k = kc + kk;
            half16 fb  = hm_ldfrag(&bsh[buf][rsel][kk + hi * 8]);
            half16 f10 = hm_ldfrag(a1p0 + k);
            half16 f20 = hm_ldfrag(a2p0 + k);
            half16 f11 = hm_ldfrag(a1p1 + k);
            half16 f21 = hm_ldfrag(a2p1 + k);
            acc1a = __builtin_amdgcn_wmma_f32_16x16x32_f16(
                false, f10, false, fb, (short)0, acc1a, false, false);
            acc2a = __builtin_amdgcn_wmma_f32_16x16x32_f16(
                false, f20, false, fb, (short)0, acc2a, false, false);
            acc1b = __builtin_amdgcn_wmma_f32_16x16x32_f16(
                false, f11, false, fb, (short)0, acc1b, false, false);
            acc2b = __builtin_amdgcn_wmma_f32_16x16x32_f16(
                false, f21, false, fb, (short)0, acc2b, false, false);
        }
    }
#undef HM_STAGE

    // C/D layout: VGPR v, lanes 0-15 -> M = v, lanes 16-31 -> M = v + 8
    float* g1 = accb;
    float* g2 = accb + (size_t)N_Q * T_PAD;
    const int col = n0 + rsel;
#pragma unroll
    for (int v = 0; v < 8; ++v) {
        int row0 = mt0 * 16 + hi * 8 + v;
        atomicAdd(&g1[(size_t)row0 * T_PAD + col], acc1a[v]);
        atomicAdd(&g2[(size_t)row0 * T_PAD + col], acc2a[v]);
        if (m1ok) {
            int row1 = mt1 * 16 + hi * 8 + v;
            atomicAdd(&g1[(size_t)row1 * T_PAD + col], acc1b[v]);
            atomicAdd(&g2[(size_t)row1 * T_PAD + col], acc2b[v]);
        }
    }
}

// ---------------------------------------------------------------------------
// Kernel 5: finalize dice + BCE mask costs into d_out
//   cost_dice = -(2*dot_sig + 1) / (max(rowsum_sig + tgtsum, 1e-6) + 1)
//   cost_mask = -(dot_diff + rowsum_log1mp) / HW
// ---------------------------------------------------------------------------
__global__ void hm_finalize(const float* __restrict__ accb,
                            const float* __restrict__ sumS,
                            const float* __restrict__ sumL,
                            const float* __restrict__ tsum,
                            float* __restrict__ out) {
    int idx = blockIdx.x * blockDim.x + threadIdx.x;
    if (idx >= N_Q * T_TGT) return;
    int nq = idx / T_TGT, t = idx - nq * T_TGT;
    float d1 = accb[(size_t)nq * T_PAD + t];
    float d2 = accb[(size_t)N_Q * T_PAD + (size_t)nq * T_PAD + t];
    float denom = sumS[nq] + tsum[t];
    float cdice = -(2.f * d1 + 1.f) / (fmaxf(denom, 1e-6f) + 1.f);
    float cmask = -(d2 + sumL[nq]) / (float)HW;
    out[idx] += 2.0f * cmask + 2.0f * cdice;
}

// ---------------------------------------------------------------------------
extern "C" void kernel_launch(void* const* d_in, const int* in_sizes, int n_in,
                              void* d_out, int out_size, void* d_ws, size_t ws_size,
                              hipStream_t stream) {
    const float* pred_logits = (const float*)d_in[0];
    const float* pred_boxes  = (const float*)d_in[1];
    const float* pred_masks  = (const float*)d_in[2];
    const int*   tgt_ids     = (const int*)d_in[3];
    const float* tgt_bbox    = (const float*)d_in[4];
    const float* tgt_masks   = (const float*)d_in[5];
    float* out = (float*)d_out;

    // workspace layout (256B-aligned partitions), ~135 MB total
    char* ws = (char*)d_ws;
    size_t offA1  = 0;
    size_t offA2  = offA1 + (size_t)N_Q * HW * 2;       // 61,440,000
    size_t offB   = offA2 + (size_t)N_Q * HW * 2;       // +61,440,000
    size_t offS1  = offB  + (size_t)T_PAD * HW * 2;     // +10,649,600
    size_t offSL  = offS1 + 4864;
    size_t offTS  = offSL + 4864;
    size_t offAcc = offTS + 1024;

    _Float16* A1 = (_Float16*)(ws + offA1);
    _Float16* A2 = (_Float16*)(ws + offA2);
    _Float16* Bm = (_Float16*)(ws + offB);
    float* sumS  = (float*)(ws + offS1);
    float* sumL  = (float*)(ws + offSL);
    float* tsum  = (float*)(ws + offTS);
    float* accb  = (float*)(ws + offAcc);

    // zero the K-split partial accumulators (graph-capture safe)
    hipMemsetAsync(accb, 0, 2 * (size_t)N_Q * T_PAD * sizeof(float), stream);

    hm_tgt_prep<<<T_PAD, 256, 0, stream>>>(tgt_masks, Bm, tsum);
    hm_pred_prep<<<N_Q, 256, 0, stream>>>(pred_masks, A1, A2, sumS, sumL);
    hm_base_cost<<<(N_Q * T_TGT + 255) / 256, 256, 0, stream>>>(
        pred_logits, pred_boxes, tgt_ids, tgt_bbox, out);

    dim3 gg(MBLK, NT, NKCH);
    hm_mask_gemm<<<gg, 32, 0, stream>>>(A1, A2, Bm, accb);

    hm_finalize<<<(N_Q * T_TGT + 255) / 256, 256, 0, stream>>>(
        accb, sumS, sumL, tsum, out);
}
